// MultiLayerLSTM_74612171866303
// MI455X (gfx1250) — matrix operand order repaired and checked by reading, not compiled, not run on hardware
//
#include <hip/hip_runtime.h>
#include <hip/hip_bf16.h>

#define Bsz 64
#define Tsz 2048
#define Isz 128
#define Hsz 128
#define NWG 8            // 8 persistent workgroups, each owns 16 hidden columns
#define NTHR 256         // 8 waves of 32
#define APITCH 264       // 256 + 8 halfs pad -> 528B row pitch (16B-aligned, bank-spread)
#define GPITCH 68        // 64 + 4 floats pad
#define BTH ((size_t)Bsz * Tsz * Hsz)
#define BH (Bsz * Hsz)

typedef __attribute__((ext_vector_type(16))) _Float16 v16h;
typedef __attribute__((ext_vector_type(8)))  _Float16 v8h;
typedef __attribute__((ext_vector_type(4)))  _Float16 v4h;
typedef __attribute__((ext_vector_type(8)))  float    v8f;
typedef __attribute__((ext_vector_type(4)))  float    v4f;

union V16U { v16h v; v8h h[2]; };

__device__ __forceinline__ float sigf(float v) { return 1.0f / (1.0f + __expf(-v)); }

// A-matrix fragment (16x32 f16, M x K) from row-major LDS [rows][APITCH].
// Lane l: row = l&15 ; K elems {0..7,16..23} (lanes 0-15) or {8..15,24..31} (lanes 16-31).
__device__ __forceinline__ v16h frag_a(const _Float16* __restrict__ base, int rowBase,
                                       int kBase, int lane) {
  const int hi = (lane >> 4) & 1;
  const _Float16* p = base + (rowBase + (lane & 15)) * APITCH + kBase + hi * 8;
  V16U u;
  u.h[0] = *(const v8h*)(p);
  u.h[1] = *(const v8h*)(p + 16);
  return u.v;
}

// B-matrix fragment (32x16 f16, K x N) where B = W^T and W is row-major [rows][APITCH].
// Lane l holds B column n = l&15 == W row (rowBase + l&15), K elems {0..7,16..23} (lanes 0-15) or {8..15,24..31} (lanes 16-31).
__device__ __forceinline__ v16h frag_b(const _Float16* __restrict__ base, int rowBase,
                                       int kBase, int lane) {
  const int hi = (lane >> 4) & 1;
  const _Float16* p = base + (rowBase + (lane & 15)) * APITCH + kBase + hi * 8;
  V16U u;
  u.h[0] = *(const v8h*)(p);
  u.h[1] = *(const v8h*)(p + 16);
  return u.v;
}

// One fused-gate GEMM slice: G[64 x 64] = A[64 x 256] * W^T[256 x 64] + bias.
// 16 output tiles (4 M x 4 N); wave w owns mtile = w>>1, ntiles {2*(w&1), 2*(w&1)+1}.
__device__ __forceinline__ void gemm_slice(const _Float16* __restrict__ A,
                                           const _Float16* __restrict__ W,
                                           const float* __restrict__ bS,
                                           float* __restrict__ G, int lane, int wave) {
  const int mtile = wave >> 1;
  const int nt0   = (wave & 1) * 2;
  const float bias0 = bS[nt0 * 16 + (lane & 15)];
  const float bias1 = bS[nt0 * 16 + 16 + (lane & 15)];
  v8f c0, c1;
#pragma unroll
  for (int i = 0; i < 8; ++i) { c0[i] = bias0; c1[i] = bias1; }
#pragma unroll
  for (int kc = 0; kc < 8; ++kc) {
    v16h a  = frag_a(A, mtile * 16, kc * 32, lane);
    v16h b0 = frag_b(W, nt0 * 16,      kc * 32, lane);
    v16h b1 = frag_b(W, nt0 * 16 + 16, kc * 32, lane);
    c0 = __builtin_amdgcn_wmma_f32_16x16x32_f16(false, a, false, b0, (short)0, c0, false, false);
    c1 = __builtin_amdgcn_wmma_f32_16x16x32_f16(false, a, false, b1, (short)0, c1, false, false);
  }
  // C/D layout: VGPR v, lane l -> M = v + 8*(l>=16), N = l&15
  const int hi   = (lane >> 4) & 1;
  const int mrow = mtile * 16 + hi * 8;
  const int nc   = nt0 * 16 + (lane & 15);
#pragma unroll
  for (int v = 0; v < 8; ++v) {
    G[(mrow + v) * GPITCH + nc]      = c0[v];
    G[(mrow + v) * GPITCH + nc + 16] = c1[v];
  }
}

// Device-wide sense-reversing barrier (works across graph replays; state re-inited per call).
__device__ __forceinline__ void grid_barrier(unsigned* cnt, unsigned* gen) {
  __syncthreads();
  if (threadIdx.x == 0) {
    __threadfence();
    volatile unsigned* vgen = gen;
    unsigned g = *vgen;
    unsigned old = atomicAdd(cnt, 1u);
    if (old == NWG - 1) {
      atomicExch(cnt, 0u);
      __threadfence();
      atomicAdd(gen, 1u);
    } else {
      while (*vgen == g) { __builtin_amdgcn_s_sleep(2); }
    }
    __threadfence();
  }
  __syncthreads();
}

__global__ void lstm_init_ws(_Float16* hsh0, _Float16* hsh1, unsigned* bar) {
  int tid = blockIdx.x * blockDim.x + threadIdx.x;
  if (tid < 2) bar[tid] = 0u;
  for (int f = tid; f < 2 * BH; f += gridDim.x * blockDim.x) {
    hsh0[f] = (_Float16)0.0f;
    hsh1[f] = (_Float16)0.0f;
  }
}

__global__ __launch_bounds__(NTHR, 1)
void lstm_persistent(const float* __restrict__ x,
                     const float* __restrict__ Wx0, const float* __restrict__ Wh0,
                     const float* __restrict__ b0,
                     const float* __restrict__ Wx1, const float* __restrict__ Wh1,
                     const float* __restrict__ b1,
                     float* __restrict__ out,
                     _Float16* __restrict__ hsh0, _Float16* __restrict__ hsh1,
                     unsigned* __restrict__ bar) {
  extern __shared__ char smem[];
  _Float16* A   = (_Float16*)smem;              // [64][APITCH]  activations [x|h] f16
  _Float16* W0  = A  + 64 * APITCH;             // [64][APITCH]  [Wx0|Wh0] slice f16
  _Float16* W1  = W0 + 64 * APITCH;             // [64][APITCH]  [Wx1|Wh1] slice f16
  float*    G   = (float*)(W1 + 64 * APITCH);   // [64][GPITCH]  fp32 gate staging
  float*    bS0 = G + 64 * GPITCH;              // [64] bias slice layer0
  float*    bS1 = bS0 + 64;                     // [64] bias slice layer1

  const int tid  = threadIdx.x;
  const int lane = tid & 31;
  const int wave = tid >> 5;
  const int wg   = blockIdx.x;                  // owns H cols [wg*16, wg*16+16)

  // ---- Load weight slices fp32 -> f16 into LDS (once). Local gate-row r -> global
  //      row grow = (r>>4)*128 + wg*16 + (r&15)  (gate order i,f,g,o preserved).
  for (int f = tid; f < 64 * 128; f += NTHR) {
    int r = f >> 7, k = f & 127;
    int grow = (r >> 4) * Hsz + wg * 16 + (r & 15);
    W0[r * APITCH + k]       = (_Float16)Wx0[grow * Isz + k];
    W0[r * APITCH + 128 + k] = (_Float16)Wh0[grow * Hsz + k];
    W1[r * APITCH + k]       = (_Float16)Wx1[grow * Hsz + k];
    W1[r * APITCH + 128 + k] = (_Float16)Wh1[grow * Hsz + k];
  }
  for (int r = tid; r < 64; r += NTHR) {
    int grow = (r >> 4) * Hsz + wg * 16 + (r & 15);
    bS0[r] = b0[grow];
    bS1[r] = b1[grow];
  }

  // Per-thread cell state: 4 consecutive (b, j) elements of this WG's 64x16 h-slice.
  const int e0 = tid * 4;
  const int eb = e0 >> 4;        // batch row 0..63
  const int ej = e0 & 15;        // local column 0,4,8,12
  float c0v[4] = {0.f, 0.f, 0.f, 0.f};
  float c1v[4] = {0.f, 0.f, 0.f, 0.f};

  __syncthreads();

  for (int t = 0; t < Tsz; ++t) {
    const int pr = t & 1;        // read parity
    const int pw = pr ^ 1;       // write parity

    // ---- A = [x_t | h0_prev]  (vectorized 8-wide, f16)
#pragma unroll
    for (int rep = 0; rep < 4; ++rep) {
      int f = rep * 2048 + tid * 8;        // 8192 elems, 8-contiguous per thread
      int bb = f >> 7, cc = f & 127;
      const float* xp = x + (size_t)bb * Tsz * Isz + (size_t)t * Isz + cc;
      v4f x0 = *(const v4f*)(xp);
      v4f x1 = *(const v4f*)(xp + 4);
      v8h hx;
#pragma unroll
      for (int i = 0; i < 4; ++i) { hx[i] = (_Float16)x0[i]; hx[4 + i] = (_Float16)x1[i]; }
      *(v8h*)&A[bb * APITCH + cc]       = hx;
      *(v8h*)&A[bb * APITCH + 128 + cc] = *(const v8h*)&hsh0[pr * BH + f];
    }
    __syncthreads();

    gemm_slice(A, W0, bS0, G, lane, wave);
    __syncthreads();

    // ---- layer-0 elementwise; publish h0_new slice (f16) to global for all WGs
    {
      v4h hv;
#pragma unroll
      for (int q = 0; q < 4; ++q) {
        int j = ej + q;
        float gi = G[eb * GPITCH + j];
        float gf = G[eb * GPITCH + 16 + j];
        float gg = G[eb * GPITCH + 32 + j];
        float go = G[eb * GPITCH + 48 + j];
        float cv = sigf(gf) * c0v[q] + sigf(gi) * tanhf(gg);
        c0v[q] = cv;
        float h = sigf(go) * tanhf(cv);
        hv[q] = (_Float16)h;
        if (t == Tsz - 1) {
          out[BTH + (size_t)eb * Hsz + wg * 16 + j] = h;                    // hT[0]
          out[BTH + 2 * (size_t)BH + (size_t)eb * Hsz + wg * 16 + j] = cv;  // cT[0]
        }
      }
      *(v4h*)&hsh0[pw * BH + eb * Hsz + wg * 16 + ej] = hv;
    }

    grid_barrier(bar, bar + 1);   // single device sync per timestep (parity double-buffer)

    // ---- A = [h0_new | h1_prev]
#pragma unroll
    for (int rep = 0; rep < 4; ++rep) {
      int f = rep * 2048 + tid * 8;
      int bb = f >> 7, cc = f & 127;
      *(v8h*)&A[bb * APITCH + cc]       = *(const v8h*)&hsh0[pw * BH + f];
      *(v8h*)&A[bb * APITCH + 128 + cc] = *(const v8h*)&hsh1[pr * BH + f];
    }
    __syncthreads();

    gemm_slice(A, W1, bS1, G, lane, wave);
    __syncthreads();

    // ---- layer-1 elementwise; write output row + publish h1 slice
    {
      v4h hv; v4f ho;
#pragma unroll
      for (int q = 0; q < 4; ++q) {
        int j = ej + q;
        float gi = G[eb * GPITCH + j];
        float gf = G[eb * GPITCH + 16 + j];
        float gg = G[eb * GPITCH + 32 + j];
        float go = G[eb * GPITCH + 48 + j];
        float cv = sigf(gf) * c1v[q] + sigf(gi) * tanhf(gg);
        c1v[q] = cv;
        float h = sigf(go) * tanhf(cv);
        hv[q] = (_Float16)h;
        ho[q] = h;
        if (t == Tsz - 1) {
          out[BTH + (size_t)BH + (size_t)eb * Hsz + wg * 16 + j] = h;       // hT[1]
          out[BTH + 3 * (size_t)BH + (size_t)eb * Hsz + wg * 16 + j] = cv;  // cT[1]
        }
      }
      *(v4h*)&hsh1[pw * BH + eb * Hsz + wg * 16 + ej] = hv;
      *(v4f*)&out[(size_t)eb * Tsz * Hsz + (size_t)t * Hsz + wg * 16 + ej] = ho;
    }
  }
}

extern "C" void kernel_launch(void* const* d_in, const int* in_sizes, int n_in,
                              void* d_out, int out_size, void* d_ws, size_t ws_size,
                              hipStream_t stream) {
  const float* x   = (const float*)d_in[0];
  const float* Wx0 = (const float*)d_in[1];
  const float* Wh0 = (const float*)d_in[2];
  const float* b0  = (const float*)d_in[3];
  const float* Wx1 = (const float*)d_in[4];
  const float* Wh1 = (const float*)d_in[5];
  const float* b1  = (const float*)d_in[6];
  float* out = (float*)d_out;

  char* ws = (char*)d_ws;
  unsigned* bar   = (unsigned*)ws;                   // 2 uints (cnt, gen)
  _Float16* hsh0  = (_Float16*)(ws + 256);           // [2][64][128] f16 h0 double-buffer
  _Float16* hsh1  = hsh0 + 2 * BH;                   // [2][64][128] f16 h1 double-buffer

  lstm_init_ws<<<8, 256, 0, stream>>>(hsh0, hsh1, bar);

  const size_t smem = (size_t)(3 * 64 * APITCH) * sizeof(_Float16)   // A, W0, W1
                    + (size_t)(64 * GPITCH) * sizeof(float)          // G
                    + 128 * sizeof(float);                           // bS0, bS1
  lstm_persistent<<<NWG, NTHR, smem, stream>>>(x, Wx0, Wh0, b0, Wx1, Wh1, b1,
                                               out, hsh0, hsh1, bar);
}
